// DynamicBlock_59734405152788
// MI455X (gfx1250) — matrix operand order repaired and checked by reading, not compiled, not run on hardware
//
#include <hip/hip_runtime.h>
#include <cstdint>

// ---------------- problem constants (match reference) ----------------
#define NTOK 2048   // N selected tokens
#define DIM  1024   // D hidden
#define NH   16     // heads
#define HD   64     // head dim
#define FF   4096   // MLP inner
#define BB   4      // batch
#define SS   2048   // seq len

typedef __attribute__((ext_vector_type(16))) _Float16 v16h;
typedef __attribute__((ext_vector_type(8)))  float    v8f;

__device__ __forceinline__ v8f vzero8() {
  v8f z;
#pragma unroll
  for (int i = 0; i < 8; ++i) z[i] = 0.0f;
  return z;
}

__device__ __forceinline__ v8f wmma16(v16h a, v16h b, v8f c) {
  // v_wmma_f32_16x16x32_f16: D = A(16x32) x B(32x16) + C(16x16 f32)
  return __builtin_amdgcn_wmma_f32_16x16x32_f16(false, a, false, b, (short)0, c,
                                                false, false);
}

// ---- CDNA5 async global->LDS copy (16B per lane), tracked by ASYNCcnt ----
// Generic pointers into __shared__ carry the wave-relative LDS offset in the
// low 32 bits (flat LDS aperture maps addr[31:0] -> LDS).
__device__ __forceinline__ void async_g2l_b128(void* lds, const void* g) {
  asm volatile("global_load_async_to_lds_b128 %0, %1, off"
               :
               : "v"((unsigned)(uintptr_t)lds), "v"(g)
               : "memory");
}
#define ASYNC_WAIT_0() asm volatile("s_wait_asynccnt 0" ::: "memory")
#define ASYNC_WAIT_4() asm volatile("s_wait_asynccnt 4" ::: "memory")

// A fragment (16x32 f16), this code's k order: lane L holds row L%16; K-half by L/16.
// Lane half h holds k = 16h..16h+15, contiguous in LDS -> 2x ds_load_b128 per lane.
__device__ __forceinline__ v16h frag_a(const _Float16* __restrict__ p, int ld,
                                       int lane) {
  const int row = lane & 15;
  const int kh = (lane >> 4) << 4;
  const _Float16* rp = p + row * ld;
  v16h a;
#pragma unroll
  for (int i = 0; i < 4; ++i) {
    a[2 * i]     = rp[kh + 2 * i];
    a[2 * i + 1] = rp[kh + 2 * i + 1];
    a[2 * i + 8] = rp[8 + kh + 2 * i];
    a[2 * i + 9] = rp[8 + kh + 2 * i + 1];
  }
  return a;
}

// B fragment from pre-swizzled LDS: 16 contiguous halves per lane
// (2x ds_load_b128, no cross-lane packing).
__device__ __forceinline__ v16h frag_b_lin(const _Float16* __restrict__ p,
                                           int lane) {
  const _Float16* rp = p + lane * 16;
  v16h b;
#pragma unroll
  for (int j = 0; j < 16; ++j) b[j] = rp[j];
  return b;
}

// B fragment from a row-major [N][K] tile (transposed operand: contiguous
// per lane -> 2x ds_load_b128). Used for Q·K^T and P·V (with V^T tile).
__device__ __forceinline__ v16h frag_bt(const _Float16* __restrict__ p, int ld,
                                        int lane) {
  const int col = lane & 15;
  const int kb = (lane >> 4) << 4;
  const _Float16* rp = p + col * ld + kb;
  v16h b;
#pragma unroll
  for (int j = 0; j < 16; ++j) b[j] = rp[j];
  return b;
}

// ---------------- weight conversion + B-fragment swizzle ----------------
// Output layout: for (kblk = K/32, nt = N/16): 32 lanes x 16 halves where
// lane L (col = L&15, khalf = L>>4), element j = W[kblk*32+khalf*16+j][nt*16+col].
// Flat: swz[(((kblk*(N/16))+nt)*32 + L)*16 + j]. GEMM B staging then copies a
// contiguous 8KB block per 32x128 tile and lanes read contiguous 32B.
__global__ __launch_bounds__(256) void k_convert_swzB(
    const float* __restrict__ W, _Float16* __restrict__ swz, int K, int N) {
  (void)K;
  __shared__ float tle[32 * 132];  // 32 x 128 (+4 pad) f32
  const int tid = threadIdx.x;
  const int k0 = blockIdx.y * 32;
  const int n0 = blockIdx.x * 128;
  {  // coalesced 32x128 tile load
    const int idx = tid * 16;
    const int r = idx >> 7, c = idx & 127;
    const float* src = W + (size_t)(k0 + r) * N + n0 + c;
#pragma unroll
    for (int u = 0; u < 16; u += 4)
      *(float4*)&tle[r * 132 + c + u] = *(const float4*)(src + u);
  }
  __syncthreads();
  // swizzled write: 8 subtiles x 512 halves, fully contiguous
  _Float16* dst =
      swz + ((((size_t)(k0 >> 5)) * (N >> 4) + (n0 >> 4)) << 9) + (size_t)tid * 16;
  const int L = tid & 31;
  const int st = tid >> 5;  // subtile 0..7
  const int col = (L & 15) + st * 16;
  const int krow = (L >> 4) << 4;
#pragma unroll
  for (int j = 0; j < 16; ++j) dst[j] = (_Float16)tle[(krow + j) * 132 + col];
}

// ---------------- generic WMMA GEMM: C = A16[M,K] @ swzB[K,N] (+bias)(+res) -
// block = 256 threads (8 waves), 128x128 C tile, K-step 32, async double-
// buffered LDS staging, branch-free templated epilogue.
template <int BIAS, int RES>
__global__ __launch_bounds__(256) void k_gemm_f16(
    const _Float16* __restrict__ A, const _Float16* __restrict__ Bswz,
    const float* __restrict__ bias, const float* __restrict__ res,
    float* __restrict__ C, int M, int Nn, int K) {
  (void)M;
  constexpr int LDA = 40;  // 32 + pad
  __shared__ _Float16 As[2][128 * LDA];
  __shared__ _Float16 Bs[2][4096];  // 8 subtiles x 512 halves, linear

  const int tid = threadIdx.x;
  const int lane = tid & 31;
  const int w = tid >> 5;
  const int m0 = blockIdx.y * 128;
  const int n0 = blockIdx.x * 128;
  const int wr = w >> 1;  // 0..3 : 32-row band
  const int wc = w & 1;   // 0..1 : 64-col band
  const int rr8 = (lane >> 4) << 3;
  const int ccol = lane & 15;

  const int aidx = tid * 16;
  const int ar = aidx >> 5, ac = aidx & 31;
  const size_t ntiles = (size_t)(Nn >> 4);

  auto stage = [&](int buf, int k0) {
    const _Float16* sa = A + (size_t)(m0 + ar) * K + k0 + ac;
    _Float16* da = &As[buf][ar * LDA + ac];
    async_g2l_b128(da, sa);
    async_g2l_b128(da + 8, sa + 8);
    const _Float16* sb =
        Bswz + ((((size_t)(k0 >> 5)) * ntiles + (n0 >> 4)) << 9) + aidx;
    _Float16* db = &Bs[buf][aidx];
    async_g2l_b128(db, sb);
    async_g2l_b128(db + 8, sb + 8);
  };

  v8f acc[2][4];
#pragma unroll
  for (int i = 0; i < 2; ++i)
#pragma unroll
    for (int j = 0; j < 4; ++j) acc[i][j] = vzero8();

  stage(0, 0);
  for (int k0 = 0; k0 < K; k0 += 32) {
    const int buf = (k0 >> 5) & 1;
    const bool have_next = (k0 + 32) < K;
    if (have_next) {
      stage(buf ^ 1, k0 + 32);
      ASYNC_WAIT_4();  // retire current stage, leave next in flight
    } else {
      ASYNC_WAIT_0();
    }
    __syncthreads();

    v16h af[2], bf[4];
#pragma unroll
    for (int i = 0; i < 2; ++i)
      af[i] = frag_a(&As[buf][(wr * 32 + i * 16) * LDA], LDA, lane);
#pragma unroll
    for (int j = 0; j < 4; ++j)
      bf[j] = frag_b_lin(&Bs[buf][(wc * 4 + j) * 512], lane);
#pragma unroll
    for (int i = 0; i < 2; ++i)
#pragma unroll
      for (int j = 0; j < 4; ++j) acc[i][j] = wmma16(af[i], bf[j], acc[i][j]);
    __syncthreads();  // protect buf from overwrite two stages ahead
  }

#pragma unroll
  for (int i = 0; i < 2; ++i) {
#pragma unroll
    for (int j = 0; j < 4; ++j) {
#pragma unroll
      for (int r = 0; r < 8; ++r) {
        const int row = m0 + wr * 32 + i * 16 + rr8 + r;
        const int col = n0 + wc * 64 + j * 16 + ccol;
        float v = acc[i][j][r];
        if constexpr (BIAS) v += bias[col];
        if constexpr (RES) v += res[(size_t)row * Nn + col];
        C[(size_t)row * Nn + col] = v;
      }
    }
  }
}

// ---------------- elementwise / reduction kernels ----------------
__global__ __launch_bounds__(256) void k_gather(
    const float* __restrict__ hidden, const int* __restrict__ bi,
    const int* __restrict__ ti, const float* __restrict__ cosT,
    const float* __restrict__ sinT, float* __restrict__ sel,
    float* __restrict__ cg, float* __restrict__ sg) {
  const int n = blockIdx.x;
  const int b = bi[n], t = ti[n];
  const float* src = hidden + ((size_t)b * SS + t) * DIM;
  float* dst = sel + (size_t)n * DIM;
  for (int i = threadIdx.x; i < DIM; i += 256) dst[i] = src[i];
  if (threadIdx.x < HD) {
    const size_t base = ((size_t)b * SS + t) * HD + threadIdx.x;
    cg[n * HD + threadIdx.x] = cosT[base];
    sg[n * HD + threadIdx.x] = sinT[base];
  }
}

__global__ __launch_bounds__(256) void k_rmsnorm(const float* __restrict__ x,
                                                 const float* __restrict__ w,
                                                 _Float16* __restrict__ y) {
  __shared__ float red[256];
  const int n = blockIdx.x;
  const float* xr = x + (size_t)n * DIM;
  float s = 0.f;
  for (int i = threadIdx.x; i < DIM; i += 256) {
    const float v = xr[i];
    s += v * v;
  }
  red[threadIdx.x] = s;
  __syncthreads();
  for (int off = 128; off > 0; off >>= 1) {
    if (threadIdx.x < off) red[threadIdx.x] += red[threadIdx.x + off];
    __syncthreads();
  }
  const float rstd = rsqrtf(red[0] / (float)DIM + 1e-6f);
  for (int i = threadIdx.x; i < DIM; i += 256)
    y[(size_t)n * DIM + i] = (_Float16)(xr[i] * rstd * w[i]);
}

// RoPE on q,k (f32 in) + repack to head-major f16 [H][N][HD]; v converted too.
__global__ __launch_bounds__(256) void k_rope(
    const float* __restrict__ q, const float* __restrict__ k,
    const float* __restrict__ v, const float* __restrict__ cg,
    const float* __restrict__ sg, _Float16* __restrict__ q16,
    _Float16* __restrict__ k16, _Float16* __restrict__ v16) {
  const int n = blockIdx.x;
  for (int idx = threadIdx.x; idx < NH * HD; idx += 256) {
    const int h = idx >> 6, d = idx & 63;
    const float c = cg[n * HD + d], s = sg[n * HD + d];
    const size_t gi = (size_t)n * (NH * HD) + idx;
    const float qv = q[gi], kv = k[gi];
    float qo, ko;
    if (d < 32) {  // rotate_half: first half gets -x2*s
      qo = qv * c - q[gi + 32] * s;
      ko = kv * c - k[gi + 32] * s;
    } else {       // second half gets +x1*s
      qo = qv * c + q[gi - 32] * s;
      ko = kv * c + k[gi - 32] * s;
    }
    const size_t hi = ((size_t)h * NTOK + n) * HD + d;
    q16[hi] = (_Float16)qo;
    k16[hi] = (_Float16)ko;
    v16[hi] = (_Float16)v[gi];
  }
}

// LDS-tiled transpose: v16 [H][N][HD] -> vT [H][HD][N] so the attention P·V
// B-operand reads contiguous per lane.
__global__ __launch_bounds__(256) void k_transpose_v(
    const _Float16* __restrict__ v16, _Float16* __restrict__ vT) {
  __shared__ _Float16 t[64 * 65];
  const int tid = threadIdx.x;
  const int n0 = blockIdx.x * 64;
  const int h = blockIdx.y;
  const int idx = tid * 16;
  const int r = idx >> 6, c = idx & 63;
  {
    const _Float16* src = v16 + ((size_t)h * NTOK + n0 + r) * HD + c;
#pragma unroll
    for (int u = 0; u < 16; ++u) t[r * 65 + c + u] = src[u];  // t[n][d]
  }
  __syncthreads();
  {
    _Float16* dst = vT + ((size_t)h * HD + r) * NTOK + n0 + c;
#pragma unroll
    for (int u = 0; u < 16; ++u) dst[u] = t[(c + u) * 65 + r];  // v[n][d]->vT[d][n]
  }
}

// ---------------- flash attention (causal) ----------------
// grid = (NTOK/64, NH); block = 256 (8 waves). 64-query x 64-key tiles.
__global__ __launch_bounds__(256) void k_attn(const _Float16* __restrict__ q16,
                                              const _Float16* __restrict__ k16,
                                              const _Float16* __restrict__ vT16,
                                              _Float16* __restrict__ ctx16) {
  constexpr int LT = 72;  // f16 tile leading dim (64 + pad)
  constexpr int LS = 65;  // f32 score leading dim
  __shared__ _Float16 Qs[64 * LT];
  __shared__ _Float16 Ks[64 * LT];
  __shared__ _Float16 VsT[64 * LT];  // [d][token]
  __shared__ _Float16 Ps[64 * LT];
  __shared__ float Sf[64 * LS];
  __shared__ float mrow[64], lrow[64], arow[64], mnew[64];
  __shared__ float segM[4][64], segL[4][64];

  const int tid = threadIdx.x;
  const int lane = tid & 31;
  const int w = tid >> 5;
  const int h = blockIdx.y;
  const int q0 = blockIdx.x * 64;
  const int ti_ = w >> 1;       // 0..3 row tile
  const int tj0 = (w & 1) * 2;  // col tiles tj0, tj0+1
  const int rr8 = (lane >> 4) << 3;
  const int ccol = lane & 15;

  const _Float16* qh = q16 + (size_t)h * NTOK * HD;
  const _Float16* kh = k16 + (size_t)h * NTOK * HD;
  const _Float16* vth = vT16 + (size_t)h * HD * NTOK;  // row d, stride NTOK

  // staging geometry: each thread moves 16 contiguous halves of one tile row
  const int sidx = tid * 16;
  const int sr = sidx >> 6, sc = sidx & 63;

  {  // stage Q tile once (async)
    const _Float16* src = qh + (size_t)(q0 + sr) * HD + sc;
    _Float16* dst = Qs + sr * LT + sc;
    async_g2l_b128(dst, src);
    async_g2l_b128(dst + 8, src + 8);
  }
  if (tid < 64) {
    mrow[tid] = -3.0e38f;
    lrow[tid] = 0.f;
  }

  v8f oacc[2];
  oacc[0] = vzero8();
  oacc[1] = vzero8();

  for (int kstart = 0; kstart <= q0; kstart += 64) {
    {  // stage K tile rows=token, V^T tile rows=d (both contiguous, async)
      const _Float16* sk = kh + (size_t)(kstart + sr) * HD + sc;
      _Float16* dk = Ks + sr * LT + sc;
      async_g2l_b128(dk, sk);
      async_g2l_b128(dk + 8, sk + 8);
      const _Float16* sv = vth + (size_t)sr * NTOK + kstart + sc;
      _Float16* dv = VsT + sr * LT + sc;
      async_g2l_b128(dv, sv);
      async_g2l_b128(dv + 8, sv + 8);
    }
    ASYNC_WAIT_0();
    __syncthreads();

    // S = Q·K^T * (1/sqrt(HD)) with causal mask
#pragma unroll
    for (int jj = 0; jj < 2; ++jj) {
      const int tj = tj0 + jj;
      v8f sacc = vzero8();
#pragma unroll
      for (int kk = 0; kk < HD; kk += 32) {
        v16h a = frag_a(Qs + ti_ * 16 * LT + kk, LT, lane);
        v16h b = frag_bt(Ks + tj * 16 * LT + kk, LT, lane);
        sacc = wmma16(a, b, sacc);
      }
#pragma unroll
      for (int r = 0; r < 8; ++r) {
        const int rl = ti_ * 16 + rr8 + r;
        const int cl = tj * 16 + ccol;
        float v = sacc[r] * 0.125f;  // 1/sqrt(64)
        if (kstart + cl > q0 + rl) v = -3.0e38f;
        Sf[rl * LS + cl] = v;
      }
    }
    __syncthreads();

    // online softmax: 4 segment-threads per row
    {
      const int r = tid & 63, sg_ = tid >> 6;
      const float* srow = Sf + r * LS + sg_ * 16;
      float mx = -3.0e38f;
#pragma unroll
      for (int c = 0; c < 16; ++c) mx = fmaxf(mx, srow[c]);
      segM[sg_][r] = mx;
    }
    __syncthreads();
    if (tid < 64) {
      const int r = tid;
      float mx = fmaxf(fmaxf(segM[0][r], segM[1][r]),
                       fmaxf(segM[2][r], segM[3][r]));
      mnew[r] = fmaxf(mx, mrow[r]);
    }
    __syncthreads();
    {
      const int r = tid & 63, sg_ = tid >> 6;
      const float mx = mnew[r];
      const float* srow = Sf + r * LS + sg_ * 16;
      _Float16* pr = Ps + r * LT + sg_ * 16;
      float ls = 0.f;
#pragma unroll
      for (int c = 0; c < 16; ++c) {
        const float p = expf(srow[c] - mx);
        pr[c] = (_Float16)p;
        ls += p;
      }
      segL[sg_][r] = ls;
    }
    __syncthreads();
    if (tid < 64) {
      const int r = tid;
      const float alpha = expf(mrow[r] - mnew[r]);
      lrow[r] = lrow[r] * alpha + segL[0][r] + segL[1][r] + segL[2][r] +
                segL[3][r];
      mrow[r] = mnew[r];
      arow[r] = alpha;
    }
    __syncthreads();

    // O = O*alpha + P·V  (V^T tile: frag_bt gives B[k][d] = VsT[d][k])
#pragma unroll
    for (int jj = 0; jj < 2; ++jj) {
      const int tj = tj0 + jj;
#pragma unroll
      for (int r = 0; r < 8; ++r) {
        const float t = oacc[jj][r];
        oacc[jj][r] = t * arow[ti_ * 16 + rr8 + r];
      }
#pragma unroll
      for (int kk = 0; kk < 64; kk += 32) {
        v16h a = frag_a(Ps + ti_ * 16 * LT + kk, LT, lane);
        v16h b = frag_bt(VsT + tj * 16 * LT + kk, LT, lane);
        oacc[jj] = wmma16(a, b, oacc[jj]);
      }
    }
    __syncthreads();  // before next K/V overwrite
  }

  // epilogue: O /= l; write ctx16 as [n][h*HD + d] (matches reshape(N,H*HD))
#pragma unroll
  for (int jj = 0; jj < 2; ++jj) {
    const int tj = tj0 + jj;
#pragma unroll
    for (int r = 0; r < 8; ++r) {
      const int rl = ti_ * 16 + rr8 + r;
      const int cl = tj * 16 + ccol;
      const float v = oacc[jj][r] / lrow[rl];
      ctx16[(size_t)(q0 + rl) * (NH * HD) + h * HD + cl] = (_Float16)v;
    }
  }
}

__global__ void k_silu_mul(const float* __restrict__ g,
                           const float* __restrict__ u,
                           _Float16* __restrict__ y, size_t n) {
  size_t i = blockIdx.x * 256ull + threadIdx.x;
  const size_t stride = gridDim.x * 256ull;
  for (; i < n; i += stride) {
    const float gv = g[i];
    const float sv = gv / (1.f + expf(-gv));
    y[i] = (_Float16)(sv * u[i]);
  }
}

__global__ void k_copy4(const float* __restrict__ src, float* __restrict__ dst,
                        size_t n4) {
  size_t i = blockIdx.x * 256ull + threadIdx.x;
  const size_t stride = gridDim.x * 256ull;
  const float4* s = (const float4*)src;
  float4* d = (float4*)dst;
  for (; i < n4; i += stride) d[i] = s[i];
}

__global__ __launch_bounds__(256) void k_scatter(
    const float* __restrict__ sel, const float* __restrict__ proc,
    const float* __restrict__ gate, const int* __restrict__ bi,
    const int* __restrict__ ti, float* __restrict__ out) {
  const int n = blockIdx.x;
  const int b = bi[n], t = ti[n];
  const float gv = gate[n];
  float* dst = out + ((size_t)b * SS + t) * DIM;
  const float* s = sel + (size_t)n * DIM;
  const float* p = proc + (size_t)n * DIM;
  for (int i = threadIdx.x; i < DIM; i += 256) {
    const float sv = s[i];
    dst[i] = sv + (p[i] - sv) * gv;  // soft-gated update
  }
}

// ---------------- launch ----------------
extern "C" void kernel_launch(void* const* d_in, const int* in_sizes, int n_in,
                              void* d_out, int out_size, void* d_ws,
                              size_t ws_size, hipStream_t stream) {
  (void)in_sizes; (void)n_in; (void)out_size; (void)ws_size;

  const float* hidden = (const float*)d_in[0];
  const int* bi = (const int*)d_in[1];
  const int* ti = (const int*)d_in[2];
  const float* gate = (const float*)d_in[3];
  const float* cosT = (const float*)d_in[4];
  const float* sinT = (const float*)d_in[5];
  const float* Wq = (const float*)d_in[6];
  const float* bq = (const float*)d_in[7];
  const float* Wk = (const float*)d_in[8];
  const float* bk = (const float*)d_in[9];
  const float* Wv = (const float*)d_in[10];
  const float* bv = (const float*)d_in[11];
  const float* Wo = (const float*)d_in[12];
  const float* Wg = (const float*)d_in[13];
  const float* Wu = (const float*)d_in[14];
  const float* Wd = (const float*)d_in[15];
  const float* ln1 = (const float*)d_in[16];
  const float* ln2 = (const float*)d_in[17];
  float* out = (float*)d_out;

  char* ws = (char*)d_ws;
  size_t off = 0;
  auto alloc = [&](size_t bytes) {
    char* p = ws + off;
    off += (bytes + 255) & ~(size_t)255;
    return p;
  };

  float* sel = (float*)alloc((size_t)NTOK * DIM * 4);
  float* cg = (float*)alloc((size_t)NTOK * HD * 4);
  float* sg = (float*)alloc((size_t)NTOK * HD * 4);
  _Float16* h16 = (_Float16*)alloc((size_t)NTOK * DIM * 2);
  float* qf = (float*)alloc((size_t)NTOK * DIM * 4);
  float* kf = (float*)alloc((size_t)NTOK * DIM * 4);
  float* vf = (float*)alloc((size_t)NTOK * DIM * 4);
  _Float16* q16 = (_Float16*)alloc((size_t)NH * NTOK * HD * 2);
  _Float16* k16 = (_Float16*)alloc((size_t)NH * NTOK * HD * 2);
  _Float16* v16b = (_Float16*)alloc((size_t)NH * NTOK * HD * 2);
  _Float16* vT16 = (_Float16*)alloc((size_t)NH * HD * NTOK * 2);
  _Float16* ctx16 = (_Float16*)alloc((size_t)NTOK * DIM * 2);
  float* h2 = (float*)alloc((size_t)NTOK * DIM * 4);
  _Float16* m16 = (_Float16*)alloc((size_t)NTOK * DIM * 2);
  float* gbuf = (float*)alloc((size_t)NTOK * FF * 4);
  float* ubuf = (float*)alloc((size_t)NTOK * FF * 4);
  _Float16* a16 = (_Float16*)alloc((size_t)NTOK * FF * 2);
  float* proc = (float*)alloc((size_t)NTOK * DIM * 4);
  _Float16* Wq16 = (_Float16*)alloc((size_t)DIM * DIM * 2);
  _Float16* Wk16 = (_Float16*)alloc((size_t)DIM * DIM * 2);
  _Float16* Wv16 = (_Float16*)alloc((size_t)DIM * DIM * 2);
  _Float16* Wo16 = (_Float16*)alloc((size_t)DIM * DIM * 2);
  _Float16* Wg16 = (_Float16*)alloc((size_t)DIM * FF * 2);
  _Float16* Wu16 = (_Float16*)alloc((size_t)DIM * FF * 2);
  _Float16* Wd16 = (_Float16*)alloc((size_t)FF * DIM * 2);

  // weight conversion + B-fragment swizzle (f32 -> f16), once per launch
  dim3 cvDD(DIM / 128, DIM / 32);
  k_convert_swzB<<<cvDD, 256, 0, stream>>>(Wq, Wq16, DIM, DIM);
  k_convert_swzB<<<cvDD, 256, 0, stream>>>(Wk, Wk16, DIM, DIM);
  k_convert_swzB<<<cvDD, 256, 0, stream>>>(Wv, Wv16, DIM, DIM);
  k_convert_swzB<<<cvDD, 256, 0, stream>>>(Wo, Wo16, DIM, DIM);
  dim3 cvDF(FF / 128, DIM / 32);
  k_convert_swzB<<<cvDF, 256, 0, stream>>>(Wg, Wg16, DIM, FF);
  k_convert_swzB<<<cvDF, 256, 0, stream>>>(Wu, Wu16, DIM, FF);
  dim3 cvFD(DIM / 128, FF / 32);
  k_convert_swzB<<<cvFD, 256, 0, stream>>>(Wd, Wd16, FF, DIM);

  // gather + input layernorm
  k_gather<<<NTOK, 256, 0, stream>>>(hidden, bi, ti, cosT, sinT, sel, cg, sg);
  k_rmsnorm<<<NTOK, 256, 0, stream>>>(sel, ln1, h16);

  // QKV projections (+bias)
  dim3 gq(DIM / 128, NTOK / 128);
  k_gemm_f16<1, 0><<<gq, 256, 0, stream>>>(h16, Wq16, bq, nullptr, qf, NTOK,
                                           DIM, DIM);
  k_gemm_f16<1, 0><<<gq, 256, 0, stream>>>(h16, Wk16, bk, nullptr, kf, NTOK,
                                           DIM, DIM);
  k_gemm_f16<1, 0><<<gq, 256, 0, stream>>>(h16, Wv16, bv, nullptr, vf, NTOK,
                                           DIM, DIM);

  // RoPE + head-major f16 repack, then V transpose for the P·V operand
  k_rope<<<NTOK, 256, 0, stream>>>(qf, kf, vf, cg, sg, q16, k16, v16b);
  dim3 gt(NTOK / 64, NH);
  k_transpose_v<<<gt, 256, 0, stream>>>(v16b, vT16);

  // causal flash attention
  dim3 ga(NTOK / 64, NH);
  k_attn<<<ga, 256, 0, stream>>>(q16, k16, vT16, ctx16);

  // O projection + residual 1
  k_gemm_f16<0, 1><<<gq, 256, 0, stream>>>(ctx16, Wo16, nullptr, sel, h2, NTOK,
                                           DIM, DIM);

  // post-attention layernorm + SwiGLU MLP + residual 2
  k_rmsnorm<<<NTOK, 256, 0, stream>>>(h2, ln2, m16);
  dim3 gm(FF / 128, NTOK / 128);
  k_gemm_f16<0, 0><<<gm, 256, 0, stream>>>(m16, Wg16, nullptr, nullptr, gbuf,
                                           NTOK, FF, DIM);
  k_gemm_f16<0, 0><<<gm, 256, 0, stream>>>(m16, Wu16, nullptr, nullptr, ubuf,
                                           NTOK, FF, DIM);
  k_silu_mul<<<2048, 256, 0, stream>>>(gbuf, ubuf, a16, (size_t)NTOK * FF);
  k_gemm_f16<0, 1><<<gq, 256, 0, stream>>>(a16, Wd16, nullptr, h2, proc, NTOK,
                                           DIM, FF);

  // write-through of untouched tokens, then gated scatter of updated tokens
  k_copy4<<<2048, 256, 0, stream>>>(hidden, out, (size_t)BB * SS * DIM / 4);
  k_scatter<<<NTOK, 256, 0, stream>>>(sel, proc, gate, bi, ti, out);
}
